// AdaptiveGraph_3418793968189
// MI455X (gfx1250) — matrix operand order repaired and checked by reading, not compiled, not run on hardware
//
#include <hip/hip_runtime.h>

typedef __attribute__((ext_vector_type(2)))  float  v2f;
typedef __attribute__((ext_vector_type(4)))  float  v4f;
typedef __attribute__((ext_vector_type(8)))  float  v8f;
typedef __attribute__((ext_vector_type(8)))  __bf16 v8bf;
typedef __attribute__((ext_vector_type(16))) __bf16 v16bf;
typedef __attribute__((ext_vector_type(4)))  unsigned int u32x4;
typedef __attribute__((ext_vector_type(8)))  int i32x8;
typedef __attribute__((ext_vector_type(4)))  int i32x4;

#define NN 8192
#define IN_DIM 256
#define HID 128

// ---------------- f32 WMMA helpers (16x16x4) ----------------
static __device__ __forceinline__ v2f load_k2(const float* __restrict__ p,
                                              int ld, int row0, int k0) {
  int l = threadIdx.x & 31;
  const float* q = p + (size_t)(row0 + (l & 15)) * ld + k0 + ((l >> 4) << 1);
  v2f r;
  r.x = q[0];
  r.y = q[1];
  return r;
}

static __device__ __forceinline__ v8f wmma_f32(v2f a, v2f b, v8f c) {
  return __builtin_amdgcn_wmma_f32_16x16x4_f32(false, a, false, b,
                                               (short)0, c, false, false);
}

// ---------------- bf16 WMMA helpers (16x16x32) ----------------
static __device__ __forceinline__ v8f wmma_bf16(v16bf a, v16bf b, v8f c) {
  return __builtin_amdgcn_wmma_f32_16x16x32_bf16(false, a, false, b,
                                                 (short)0, c, false, false);
}

// A-operand 16x32 bf16, K contiguous (ISA 7.12.2): lane l: row=row0+(l&15),
// runs of 8 at k0+8*(l>=16) and +16. Two global_load_b128.
static __device__ __forceinline__ v16bf load_a16(const __bf16* __restrict__ p,
                                                 int ld, int row0, int k0) {
  int l = threadIdx.x & 31;
  const __bf16* q = p + (size_t)(row0 + (l & 15)) * ld + k0 + ((l >> 4) << 3);
  v8bf r0 = *(const v8bf*)(q);
  v8bf r1 = *(const v8bf*)(q + 16);
  return __builtin_shufflevector(r0, r1, 0, 1, 2, 3, 4, 5, 6, 7,
                                 8, 9, 10, 11, 12, 13, 14, 15);
}

// B-operand 32x16 bf16 from a padded-row LDS tile: B[k][n] = T[n0+n][k].
// lane l: n=l&15; runs of 8 K at k0+8*(l>=16) and +16. Two ds_load_b128.
static __device__ __forceinline__ v16bf lds_b16(const __bf16* T, int ldp,
                                                int n0, int k0) {
  int l = threadIdx.x & 31;
  const __bf16* q = T + (size_t)(n0 + (l & 15)) * ldp + k0 + ((l >> 4) << 3);
  v8bf r0 = *(const v8bf*)(q);
  v8bf r1 = *(const v8bf*)(q + 16);
  return __builtin_shufflevector(r0, r1, 0, 1, 2, 3, 4, 5, 6, 7,
                                 8, 9, 10, 11, 12, 13, 14, 15);
}

// ---------------- Tensor Data Mover: 2D bf16 tile -> LDS ----------------
// D# per ISA 8.3/8.4. tile==tensor (fully in-bounds). LDS padding inserted
// every `pad_i`-coded DWORDs (one row) to skew banks: pad_a code 3 = 4 DWORDs.
static __device__ __forceinline__ void tdm_load_2d(unsigned lds_off,
                                                   const void* gptr,
                                                   unsigned tile_d0,  // elems/row
                                                   unsigned tile_d1,  // rows
                                                   unsigned stride,   // row stride (elems)
                                                   unsigned pad_i, unsigned pad_a) {
  unsigned long long ga = (unsigned long long)gptr;
  u32x4 g0;
  g0[0] = 1u;                                    // count=1, user descriptor
  g0[1] = lds_off;                               // lds_addr (bytes)
  g0[2] = (unsigned)ga;                          // global_addr[31:0]
  g0[3] = (unsigned)((ga >> 32) & 0x01ffffffull) // global_addr[56:32]
          | 0x80000000u;                         // type=2 ("image")
  i32x8 g1;
  g1[0] = (int)((1u << 16)                       // data_size=1 -> 2 bytes
                | (1u << 20)                     // pad_enable
                | (pad_i << 22) | (pad_a << 25));
  g1[1] = (int)((tile_d0 & 0xffffu) << 16);      // tensor_dim0[15:0]
  g1[2] = (int)((tile_d0 >> 16) | ((tile_d1 & 0xffffu) << 16)); // dim0 hi | dim1 lo
  g1[3] = (int)((tile_d1 >> 16) | ((tile_d0 & 0xffffu) << 16)); // dim1 hi | tile_dim0
  g1[4] = (int)(tile_d1 & 0xffffu);              // tile_dim1 (tile_dim2=0)
  g1[5] = (int)stride;                           // tensor_dim0_stride[31:0]
  g1[6] = 0;
  g1[7] = 0;
  i32x4 z4 = {0, 0, 0, 0};
#if defined(__clang_major__) && __clang_major__ >= 23
  i32x8 z8 = {0, 0, 0, 0, 0, 0, 0, 0};
  __builtin_amdgcn_tensor_load_to_lds(g0, g1, z4, z4, z8, 0);
#else
  __builtin_amdgcn_tensor_load_to_lds(g0, g1, z4, z4, 0);
#endif
}

__global__ void k_zero(float* __restrict__ s, int n) {
  int i = blockIdx.x * blockDim.x + threadIdx.x;
  if (i < n) s[i] = 0.0f;
}

// ------------- Kernel 1: Z = X @ W^T (f32 WMMA), emit Zb and Zt in bf16 -------------
__global__ void k_gemm_z(const float* __restrict__ X,
                         const float* __restrict__ W,
                         __bf16* __restrict__ Zb,    // [NN][HID]
                         __bf16* __restrict__ Zt) {  // [HID][NN]
  int wave = (blockIdx.x * blockDim.x + threadIdx.x) >> 5;
  int row0 = (wave >> 3) * 16;
  int col0 = (wave & 7) * 16;

  v8f acc = {};
  for (int k0 = 0; k0 < IN_DIM; k0 += 4) {
    v2f a = load_k2(X, IN_DIM, row0, k0);
    v2f b = load_k2(W, IN_DIM, col0, k0);
    acc = wmma_f32(a, b, acc);
  }

  int l  = threadIdx.x & 31;
  int n  = l & 15;
  int mh = (l >> 4) << 3;
#pragma unroll
  for (int j = 0; j < 8; ++j) {
    int r = row0 + j + mh;
    int c = col0 + n;
    __bf16 v = (__bf16)acc[j];
    Zb[(size_t)r * HID + c] = v;
    Zt[(size_t)c * NN + r]  = v;
  }
}

// ------- Kernel 2: A_un = relu(Z Z^T) via bf16 WMMA + TDM-staged B -------
// 1024 blocks = 64 row-blocks x 16 column chunks; 8 waves/block share one
// TDM-staged 512x128 bf16 B chunk in LDS (row stride 136 elems: +16B skew).
#define SIM_LDP 136
__global__ void k_sim(const __bf16* __restrict__ Zb,
                      float* __restrict__ A,     // unnormalized f32, in d_out
                      float* __restrict__ S) {
  extern __shared__ __bf16 Bsh[];                // 512 x 136 elems = 136 KB
  int rowblock = blockIdx.x >> 4;
  int chunk    = blockIdx.x & 15;
  int w        = threadIdx.x >> 5;
  int l        = threadIdx.x & 31;
  int row0     = (rowblock * 8 + w) * 16;

  if (w == 0) {
    // rows [chunk*512, +512) of Zb, full 128-elem rows (64 DW -> pad code 5)
    tdm_load_2d(0, Zb + (size_t)(chunk * 512) * HID, HID, 512, HID, 5, 3);
    __builtin_amdgcn_s_wait_tensorcnt(0);
  }

  v16bf arow[4];
#pragma unroll
  for (int t = 0; t < 4; ++t) arow[t] = load_a16(Zb, HID, row0, 32 * t);

  __syncthreads();                               // publish LDS tile

  float rsum[8];
#pragma unroll
  for (int j = 0; j < 8; ++j) rsum[j] = 0.0f;

  int n  = l & 15;
  int mh = (l >> 4) << 3;

  for (int ct = 0; ct < 32; ++ct) {
    v8f acc = {};
#pragma unroll
    for (int t = 0; t < 4; ++t)
      acc = wmma_bf16(arow[t], lds_b16(Bsh, SIM_LDP, ct * 16, 32 * t), acc);

    int col0 = chunk * 512 + ct * 16;
#pragma unroll
    for (int j = 0; j < 8; ++j) {
      float r = fmaxf(acc[j], 0.0f);
      rsum[j] += r;
      A[(size_t)(row0 + j + mh) * NN + col0 + n] = r;
    }
  }

#pragma unroll
  for (int j = 0; j < 8; ++j) {
    float v = rsum[j];
    v += __shfl_xor(v, 1, 32);
    v += __shfl_xor(v, 2, 32);
    v += __shfl_xor(v, 4, 32);
    v += __shfl_xor(v, 8, 32);
    if ((l & 15) == 0)
      atomicAdd(&S[row0 + j + mh], v);
  }
}

// ------- Kernel 3: partial out += A_un @ Z, normalize A in place -------
// 256 blocks = 64 row-blocks x 4 K-splits; 8 waves/block share the K-split so
// B (Zt slice, 128 x 512, row stride 520 elems) is TDM-staged once per 512-K
// stage. Single read + single write of the 256MB A.
#define OUT_LDP 520
__global__ void k_out(const __bf16* __restrict__ Zt,   // [HID][NN]
                      float* __restrict__ A,           // read unnorm / write norm
                      const float* __restrict__ S,
                      float* __restrict__ Oacc) {      // [NN][HID] zeroed
  extern __shared__ __bf16 Tsh[];                      // 128 x 520 elems = 130 KB
  int rowblock = blockIdx.x >> 2;
  int ksel     = blockIdx.x & 3;
  int w        = threadIdx.x >> 5;
  int l        = threadIdx.x & 31;
  int row0     = (rowblock * 8 + w) * 16;
  int m        = l & 15;
  int kh       = (l >> 4) << 3;

  float invA = 1.0f / (S[row0 + m] + 1e-6f);

  v8f acc[8];
#pragma unroll
  for (int c = 0; c < 8; ++c) acc[c] = v8f{};

  int kbase = ksel * (NN / 4);
  for (int s = 0; s < 4; ++s) {
    int kc = kbase + s * 512;
    __syncthreads();                             // previous stage fully consumed
    if (w == 0) {
      // Zt[0..127][kc..kc+511]: 512-elem rows (256 DW -> pad code 7)
      tdm_load_2d(0, Zt + kc, 512, HID, NN, 7, 3);
      __builtin_amdgcn_s_wait_tensorcnt(0);
    }
    __syncthreads();                             // publish LDS tile

    for (int kk = 0; kk < 512; kk += 32) {
      int k0 = kc + kk;
      float* pa = A + (size_t)(row0 + m) * NN + k0 + kh;
      v4f f0 = *(const v4f*)(pa);
      v4f f1 = *(const v4f*)(pa + 4);
      v4f f2 = *(const v4f*)(pa + 16);
      v4f f3 = *(const v4f*)(pa + 20);

      v16bf a;
#pragma unroll
      for (int i = 0; i < 4; ++i) {
        a[i]      = (__bf16)f0[i];
        a[4 + i]  = (__bf16)f1[i];
        a[8 + i]  = (__bf16)f2[i];
        a[12 + i] = (__bf16)f3[i];
      }

      *(v4f*)(pa)      = f0 * invA;              // normalized A, same pass
      *(v4f*)(pa + 4)  = f1 * invA;
      *(v4f*)(pa + 16) = f2 * invA;
      *(v4f*)(pa + 20) = f3 * invA;

#pragma unroll
      for (int c = 0; c < 8; ++c)
        acc[c] = wmma_bf16(a, lds_b16(Tsh, OUT_LDP, c * 16, kk), acc[c]);
    }
  }

  int n  = l & 15;
  int mh = (l >> 4) << 3;
#pragma unroll
  for (int j = 0; j < 8; ++j)
#pragma unroll
    for (int c = 0; c < 8; ++c)
      atomicAdd(&Oacc[(size_t)(row0 + j + mh) * HID + c * 16 + n], acc[j][c]);
}

// ------- Kernel 4: out = Oacc * 1/(rowsum+eps) -------
__global__ void k_scale(const float* __restrict__ Oacc,
                        const float* __restrict__ S,
                        float* __restrict__ out) {
  int i = blockIdx.x * blockDim.x + threadIdx.x;
  out[i] = Oacc[i] / (S[i >> 7] + 1e-6f);
}

extern "C" void kernel_launch(void* const* d_in, const int* in_sizes, int n_in,
                              void* d_out, int out_size, void* d_ws, size_t ws_size,
                              hipStream_t stream) {
  const float* X = (const float*)d_in[0];   // [8192, 256]
  const float* W = (const float*)d_in[1];   // [128, 256]

  float* out = (float*)d_out;                       // [8192, 128]
  float* A   = (float*)d_out + (size_t)NN * HID;    // [8192, 8192]

  __bf16* Zb = (__bf16*)d_ws;                       // [8192,128] bf16 (2 MB)
  __bf16* Zt = Zb + (size_t)NN * HID;               // [128,8192] bf16 (2 MB)
  float*  S  = (float*)(Zt + (size_t)NN * HID);     // [8192] row sums
  float*  Oa = S + NN;                              // [8192,128] f32 partials (4 MB)

  k_zero<<<(NN + NN * HID + 255) / 256, 256, 0, stream>>>(S, NN + NN * HID);

  k_gemm_z<<<512, 256, 0, stream>>>(X, W, Zb, Zt);

  k_sim<<<1024, 256, 512 * SIM_LDP * sizeof(__bf16), stream>>>(Zb, A, S);

  k_out<<<256, 256, 128 * OUT_LDP * sizeof(__bf16), stream>>>(Zt, A, S, Oa);

  k_scale<<<NN * HID / 256, 256, 0, stream>>>(Oa, S, out);
}
